// GRUDecoder_multistep_weather_14559939134125
// MI455X (gfx1250) — matrix operand order repaired and checked by reading, not compiled, not run on hardware
//
#include <hip/hip_runtime.h>
#include <hip/hip_bf16.h>

// ---------------------------------------------------------------------------
// GRU graph decoder (NRI-style) for MI455X / gfx1250, wave32 + WMMA bf16.
// Dominant work: edge-message MLP GEMMs (39200x512x256, 39200x256x256 per
// step) -> v_wmma_f32_16x16x32_bf16, f32 accumulate.  Software-pipelined
// fragment loads enforced with sched_barrier so the backend cannot collapse
// the prefetch into load->wait(0)->wmma.
// ---------------------------------------------------------------------------

typedef __attribute__((ext_vector_type(16))) __bf16 v16bf;
typedef __attribute__((ext_vector_type(8)))  float  v8f;

#define B_   16
#define T_   12
#define N_   50
#define E_   2450          // N*(N-1)
#define H_   256
#define ROWS_BE (B_ * E_)  // 39200
#define ROWS_BN (B_ * N_)  // 800

__device__ __forceinline__ v8f wmma_bf16(v16bf a, v16bf b, v8f c) {
  // 8 args: (neg_a, A, neg_b, B, c_mod, C, reuse_a, reuse_b)
  return __builtin_amdgcn_wmma_f32_16x16x32_bf16(false, a, false, b, (short)0, c,
                                                 false, false);
}

// Scheduling fence: nothing may cross (keeps prefetch loads above the WMMAs).
__device__ __forceinline__ void sched_fence() { __builtin_amdgcn_sched_barrier(0); }

union Frag { uint4 u[2]; v16bf v; };

// Fragment load with compile-time element offset (folds to immediate offsets).
// A (16x32 MxK): lanep = rowptr + (lane>=16 ? 8 : 0); pair at koff, koff+16.
__device__ __forceinline__ v16bf load_a_imm(const __bf16* lanep, int koff) {
  Frag f;
  f.u[0] = *(const uint4*)(lanep + koff);
  f.u[1] = *(const uint4*)(lanep + koff + 16);
  return f.v;
}
// B from pre-packed weights: 16 contiguous bf16 per lane per tile.
__device__ __forceinline__ v16bf load_b_imm(const __bf16* lanep, int eoff) {
  Frag f;
  f.u[0] = *(const uint4*)(lanep + eoff);
  f.u[1] = *(const uint4*)(lanep + eoff + 8);
  return f.v;
}

__device__ __forceinline__ float sigmoidf_(float x) {
  return 1.0f / (1.0f + __expf(-x));
}

// ---------------------------------------------------------------------------
// Prep: pack a (K x 256) f32 weight matrix (row stride 256) into the bf16
// B-fragment layout: dst[((kb*16+ct)*32+lane)*16+i] with
// k = kb*32 + (lane>=16 ? 8 : 0) + (i&7) + (i>=8 ? 16 : 0), col = ct*16 + (lane&15).
// ---------------------------------------------------------------------------
__global__ void k_pack(const float* __restrict__ src, __bf16* __restrict__ dst, int K) {
  int tid = blockIdx.x * blockDim.x + threadIdx.x;
  if (tid >= K * H_) return;
  int i    = tid & 15;
  int lane = (tid >> 4) & 31;
  int tl   = tid >> 9;          // kb*16 + ct   (N tiles = 16)
  int ct   = tl & 15;
  int kb   = tl >> 4;
  int k    = (kb << 5) + ((lane >> 4) << 3) + (i & 7) + ((i >> 3) << 4);
  int col  = (ct << 4) + (lane & 15);
  dst[tid] = (__bf16)src[k * H_ + col];
}

// Zero hidden (f32 + bf16), init prev_pred = inputs[:,0]
__global__ void k_init(const float* __restrict__ inputs, float* __restrict__ hid,
                       __bf16* __restrict__ hidb, float* __restrict__ prev) {
  int tid = blockIdx.x * blockDim.x + threadIdx.x;
  if (tid < ROWS_BN * H_) { hid[tid] = 0.0f; hidb[tid] = (__bf16)0.0f; }
  if (tid < ROWS_BN) {
    int b = tid / N_, n = tid - b * N_;
    prev[tid] = inputs[(b * T_ + 0) * N_ + n];   // F_IN == 1
  }
}

// ---------------------------------------------------------------------------
// K1: msg1 = tanh([h_send | h_recv] @ W1[0:512] + b1 + weather @ W1[512:514])
// rows = B*E, K = 512, N = 256.  Wave: 16 rows x 64 cols (4 accumulators),
// double-buffered fragment loads fenced against the WMMAs.
// ---------------------------------------------------------------------------
__global__ void k_msg1(const __bf16* __restrict__ hidb, const __bf16* __restrict__ Wp1,
                       const float* __restrict__ msg_w1, const float* __restrict__ msg_b1,
                       const float* __restrict__ weather, int t,
                       __bf16* __restrict__ msg1) {
  int lane = threadIdx.x & 31;
  int wv   = blockIdx.x * (blockDim.x >> 5) + (threadIdx.x >> 5);
  if (wv >= E_ * 4) return;                 // 2450 row tiles * 4 col strips
  int rt = wv >> 2, cs = wv & 3;
  int row = (rt << 4) + (lane & 15);        // global row (b*E + e)
  int b   = row / E_;
  int e   = row - b * E_;
  int s   = e / 49;
  int jj  = e - s * 49;
  int rn  = jj + (jj >= s ? 1 : 0);
  int hioff = (lane >> 4) << 3;
  const __bf16* ps = hidb + ((b * N_ + s)  << 8) + hioff;
  const __bf16* pr = hidb + ((b * N_ + rn) << 8) + hioff;
  const __bf16* wb = Wp1 + (cs << 11) + (lane << 4);  // cs*4*512 + lane*16

  v8f acc[4] = {};
  v16bf a0 = load_a_imm(ps, 0);
  v16bf b0[4];
#pragma unroll
  for (int c = 0; c < 4; ++c) b0[c] = load_b_imm(wb, c * 512);
#pragma unroll
  for (int kb = 0; kb < 16; ++kb) {
    v16bf a1; v16bf b1[4];
    if (kb < 15) {
      const __bf16* pn = ((kb + 1) < 8) ? ps : pr;
      a1 = load_a_imm(pn, ((kb + 1) & 7) << 5);
#pragma unroll
      for (int c = 0; c < 4; ++c)
        b1[c] = load_b_imm(wb, (kb + 1) * 8192 + c * 512);
    }
    sched_fence();                           // keep prefetch above compute
#pragma unroll
    for (int c = 0; c < 4; ++c) acc[c] = wmma_bf16(a0, b0[c], acc[c]);
    sched_fence();                           // keep compute above next prefetch
    a0 = a1;
#pragma unroll
    for (int c = 0; c < 4; ++c) b0[c] = b1[c];
  }

  int hi = lane >> 4, cl = lane & 15;
  int b0r = (rt << 4) / E_;                 // batch of first tile row
#pragma unroll
  for (int c = 0; c < 4; ++c) {
    int col = (((cs << 2) + c) << 4) + cl;
    float bw = msg_b1[H_ + col];                       // type-1 bias
    float wA = msg_w1[(514 + 512) * H_ + col];         // weather row 512 (type 1)
    float wB = msg_w1[(514 + 513) * H_ + col];         // weather row 513 (type 1)
#pragma unroll
    for (int v = 0; v < 8; ++v) {
      int r2 = (rt << 4) + v + (hi << 3);
      int b2 = b0r + (r2 >= (b0r + 1) * E_ ? 1 : 0);   // tile spans <= 2 batches
      const float* wt = weather + ((b2 * T_ + t) << 1);
      float val = acc[c][v] + bw + wt[0] * wA + wt[1] * wB;
      msg1[((long)r2 << 8) + col] = (__bf16)tanhf(val);
    }
  }
}

// ---------------------------------------------------------------------------
// K2: am = tanh(msg1 @ W2 + b2) * rel_types[:,:,1] * 0.5   (SKIP_FIRST)
// ---------------------------------------------------------------------------
__global__ void k_msg2(const __bf16* __restrict__ msg1, const __bf16* __restrict__ Wp2,
                       const float* __restrict__ msg_b2, const float* __restrict__ rel_types,
                       float* __restrict__ am) {
  int lane = threadIdx.x & 31;
  int wv   = blockIdx.x * (blockDim.x >> 5) + (threadIdx.x >> 5);
  if (wv >= E_ * 4) return;
  int rt = wv >> 2, cs = wv & 3;
  int row = (rt << 4) + (lane & 15);
  const __bf16* pa = msg1 + ((long)row << 8) + ((lane >> 4) << 3);
  const __bf16* wb = Wp2 + (cs << 11) + (lane << 4);

  v8f acc[4] = {};
  v16bf a0 = load_a_imm(pa, 0);
  v16bf b0[4];
#pragma unroll
  for (int c = 0; c < 4; ++c) b0[c] = load_b_imm(wb, c * 512);
#pragma unroll
  for (int kb = 0; kb < 8; ++kb) {
    v16bf a1; v16bf b1[4];
    if (kb < 7) {
      a1 = load_a_imm(pa, (kb + 1) << 5);
#pragma unroll
      for (int c = 0; c < 4; ++c)
        b1[c] = load_b_imm(wb, (kb + 1) * 8192 + c * 512);
    }
    sched_fence();
#pragma unroll
    for (int c = 0; c < 4; ++c) acc[c] = wmma_bf16(a0, b0[c], acc[c]);
    sched_fence();
    a0 = a1;
#pragma unroll
    for (int c = 0; c < 4; ++c) b0[c] = b1[c];
  }

  int hi = lane >> 4, cl = lane & 15;
#pragma unroll
  for (int c = 0; c < 4; ++c) {
    int col = (((cs << 2) + c) << 4) + cl;
    float bw = msg_b2[H_ + col];
#pragma unroll
    for (int v = 0; v < 8; ++v) {
      int r2 = (rt << 4) + v + (hi << 3);
      float sc = rel_types[(r2 << 1) + 1] * 0.5f;      // / EDGE_TYPES
      am[((long)r2 << 8) + col] = tanhf(acc[c][v] + bw) * sc;
    }
  }
}

// ---------------------------------------------------------------------------
// K3: agg[b,n,:] = (1/N) * sum_{s != n} am[b, e(s,n), :]  (analytic rel_rec)
// ---------------------------------------------------------------------------
__global__ void k_agg(const float* __restrict__ am, __bf16* __restrict__ aggb) {
  int tid = blockIdx.x * blockDim.x + threadIdx.x;
  if (tid >= ROWS_BN * H_) return;
  int col = tid & 255, rowbn = tid >> 8;
  int b = rowbn / N_, n = rowbn - b * N_;
  float s = 0.0f;
  for (int sx = 0; sx < N_; ++sx) {
    if (sx == n) continue;
    int e = sx * 49 + n - (n > sx ? 1 : 0);
    s += am[(((long)(b * E_ + e)) << 8) + col];
  }
  aggb[tid] = (__bf16)(s * (1.0f / (float)N_));
}

// ---------------------------------------------------------------------------
// K4a: GRU update.  ghr/ghi/ghn = agg @ w_h{r,i,n} via WMMA; input-path
// (K = F_IN + W_IN = 3) folded in scalar epilogue.  Writes hidden f32 + bf16.
// ---------------------------------------------------------------------------
__global__ void k_gru(const __bf16* __restrict__ aggb,
                      const __bf16* __restrict__ Wphr, const __bf16* __restrict__ Wphi,
                      const __bf16* __restrict__ Wphn,
                      const float* __restrict__ w_ir, const float* __restrict__ b_ir,
                      const float* __restrict__ w_ii, const float* __restrict__ b_ii,
                      const float* __restrict__ w_in, const float* __restrict__ b_in,
                      const float* __restrict__ inputs, const float* __restrict__ weather,
                      const float* __restrict__ prev, const int* __restrict__ burnp,
                      int t, float* __restrict__ hid, __bf16* __restrict__ hidb) {
  int lane = threadIdx.x & 31;
  int wv   = blockIdx.x * (blockDim.x >> 5) + (threadIdx.x >> 5);
  if (wv >= (ROWS_BN / 16) * 16) return;    // 50 row tiles * 16 col tiles
  int rt = wv >> 4, ct = wv & 15;
  int row = (rt << 4) + (lane & 15);
  const __bf16* pa = aggb + (row << 8) + ((lane >> 4) << 3);
  int wboff = (ct << 9) + (lane << 4);      // ct*512 + lane*16
  const __bf16* bRp = Wphr + wboff;
  const __bf16* bIp = Wphi + wboff;
  const __bf16* bNp = Wphn + wboff;

  v8f aR = {}, aI = {}, aN = {};
  v16bf a0 = load_a_imm(pa, 0);
  v16bf r0 = load_b_imm(bRp, 0), i0 = load_b_imm(bIp, 0), n0 = load_b_imm(bNp, 0);
#pragma unroll
  for (int kb = 0; kb < 8; ++kb) {
    v16bf a1, r1, i1, n1;
    if (kb < 7) {
      a1 = load_a_imm(pa, (kb + 1) << 5);
      r1 = load_b_imm(bRp, (kb + 1) * 8192);
      i1 = load_b_imm(bIp, (kb + 1) * 8192);
      n1 = load_b_imm(bNp, (kb + 1) * 8192);
    }
    sched_fence();
    aR = wmma_bf16(a0, r0, aR);
    aI = wmma_bf16(a0, i0, aI);
    aN = wmma_bf16(a0, n0, aN);
    sched_fence();
    a0 = a1; r0 = r1; i0 = i1; n0 = n1;
  }

  int hi = lane >> 4, cl = lane & 15;
  int col = (ct << 4) + cl;
  bool use_in = (t <= burnp[0]);
  float bir = b_ir[col], bii = b_ii[col], bin_ = b_in[col];
  float wir0 = w_ir[col], wir1 = w_ir[H_ + col], wir2 = w_ir[2 * H_ + col];
  float wii0 = w_ii[col], wii1 = w_ii[H_ + col], wii2 = w_ii[2 * H_ + col];
  float win0 = w_in[col], win1 = w_in[H_ + col], win2 = w_in[2 * H_ + col];
  int bb0 = (rt << 4) / N_;                 // node-row tile spans <= 2 batches
#pragma unroll
  for (int v = 0; v < 8; ++v) {
    int grow = (rt << 4) + v + (hi << 3);
    int bb = bb0 + (grow >= (bb0 + 1) * N_ ? 1 : 0);
    int nn = grow - bb * N_;
    float ins = use_in ? inputs[(bb * T_ + t) * N_ + nn] : prev[grow];
    const float* wt = weather + ((bb * T_ + t) << 1);
    float rg = sigmoidf_(aR[v] + bir + ins * wir0 + wt[0] * wir1 + wt[1] * wir2);
    float ig = sigmoidf_(aI[v] + bii + ins * wii0 + wt[0] * wii1 + wt[1] * wii2);
    float ng = tanhf(bin_ + ins * win0 + wt[0] * win1 + wt[1] * win2 + rg * aN[v]);
    int idx = (grow << 8) + col;
    float hn = (1.0f - ig) * ng + ig * hid[idx];
    hid[idx]  = hn;
    hidb[idx] = (__bf16)hn;
  }
}

// ---------------------------------------------------------------------------
// K4b/K4c: relu(A @ W + b).  One 16x16 output tile per wave, pipelined.
// ---------------------------------------------------------------------------
__global__ void k_mlp(const __bf16* __restrict__ A, const __bf16* __restrict__ Wp,
                      const float* __restrict__ bias, __bf16* __restrict__ out_bf,
                      float* __restrict__ out_f) {
  int lane = threadIdx.x & 31;
  int wv   = blockIdx.x * (blockDim.x >> 5) + (threadIdx.x >> 5);
  if (wv >= (ROWS_BN / 16) * 16) return;
  int rt = wv >> 4, ct = wv & 15;
  int row = (rt << 4) + (lane & 15);
  const __bf16* pa = A + (row << 8) + ((lane >> 4) << 3);
  const __bf16* wb = Wp + (ct << 9) + (lane << 4);

  v8f acc = {};
  v16bf a0 = load_a_imm(pa, 0);
  v16bf b0 = load_b_imm(wb, 0);
#pragma unroll
  for (int kb = 0; kb < 8; ++kb) {
    v16bf a1, b1;
    if (kb < 7) {
      a1 = load_a_imm(pa, (kb + 1) << 5);
      b1 = load_b_imm(wb, (kb + 1) * 8192);
    }
    sched_fence();
    acc = wmma_bf16(a0, b0, acc);
    sched_fence();
    a0 = a1; b0 = b1;
  }

  int hi = lane >> 4, cl = lane & 15;
  int col = (ct << 4) + cl;
  float bv = bias[col];
#pragma unroll
  for (int v = 0; v < 8; ++v) {
    int grow = (rt << 4) + v + (hi << 3);
    float val = acc[v] + bv;
    val = val > 0.0f ? val : 0.0f;
    int idx = (grow << 8) + col;
    if (out_bf) out_bf[idx] = (__bf16)val;
    if (out_f)  out_f[idx]  = val;
  }
}

// K4d: pred = ins + h2 @ w_o3 + b_o3; store into d_out and prev_pred.
__global__ void k_pred(const float* __restrict__ h2, const float* __restrict__ w_o3,
                       const float* __restrict__ b_o3, const float* __restrict__ inputs,
                       const int* __restrict__ burnp, int t,
                       float* __restrict__ prev, float* __restrict__ out) {
  int row = blockIdx.x * blockDim.x + threadIdx.x;
  if (row >= ROWS_BN) return;
  int bb = row / N_, nn = row - bb * N_;
  float acc = b_o3[0];
  const float* hp = h2 + ((long)row << 8);
  for (int j = 0; j < H_; ++j) acc += hp[j] * w_o3[j];
  float ins = (t <= burnp[0]) ? inputs[(bb * T_ + t) * N_ + nn] : prev[row];
  float pred = ins + acc;
  out[(bb * (T_ - 1) + t) * N_ + nn] = pred;   // (B, T-1, N, 1)
  prev[row] = pred;
}

// ---------------------------------------------------------------------------
extern "C" void kernel_launch(void* const* d_in, const int* in_sizes, int n_in,
                              void* d_out, int out_size, void* d_ws, size_t ws_size,
                              hipStream_t stream) {
  const float* inputs    = (const float*)d_in[0];
  const float* weather   = (const float*)d_in[1];
  // d_in[2]=rel_rec, d_in[3]=rel_send: one-hot, replaced by analytic indexing
  const float* rel_types = (const float*)d_in[4];
  const float* msg_w1 = (const float*)d_in[5];
  const float* msg_b1 = (const float*)d_in[6];
  const float* msg_w2 = (const float*)d_in[7];
  const float* msg_b2 = (const float*)d_in[8];
  const float* w_hr = (const float*)d_in[9];
  const float* w_hi = (const float*)d_in[10];
  const float* w_hn = (const float*)d_in[11];
  const float* w_ir = (const float*)d_in[12];
  const float* b_ir = (const float*)d_in[13];
  const float* w_ii = (const float*)d_in[14];
  const float* b_ii = (const float*)d_in[15];
  const float* w_in = (const float*)d_in[16];
  const float* b_in = (const float*)d_in[17];
  const float* w_o1 = (const float*)d_in[18];
  const float* b_o1 = (const float*)d_in[19];
  const float* w_o2 = (const float*)d_in[20];
  const float* b_o2 = (const float*)d_in[21];
  const float* w_o3 = (const float*)d_in[22];
  const float* b_o3 = (const float*)d_in[23];
  const int*   burnp = (const int*)d_in[25];  // burn_in_steps

  char* ws = (char*)d_ws;
  size_t off = 0;
  auto alloc = [&](size_t bytes) -> void* {
    void* p = ws + off;
    off = (off + bytes + 255) & ~(size_t)255;
    return p;
  };
  float*  hid   = (float*) alloc((size_t)ROWS_BN * H_ * 4);
  __bf16* hidb  = (__bf16*)alloc((size_t)ROWS_BN * H_ * 2);
  float*  prev  = (float*) alloc((size_t)ROWS_BN * 4);
  __bf16* Wp1   = (__bf16*)alloc((size_t)512 * H_ * 2);
  __bf16* Wp2   = (__bf16*)alloc((size_t)H_ * H_ * 2);
  __bf16* Wphr  = (__bf16*)alloc((size_t)H_ * H_ * 2);
  __bf16* Wphi  = (__bf16*)alloc((size_t)H_ * H_ * 2);
  __bf16* Wphn  = (__bf16*)alloc((size_t)H_ * H_ * 2);
  __bf16* Wpo1  = (__bf16*)alloc((size_t)H_ * H_ * 2);
  __bf16* Wpo2  = (__bf16*)alloc((size_t)H_ * H_ * 2);
  __bf16* msg1  = (__bf16*)alloc((size_t)ROWS_BE * H_ * 2);
  float*  am    = (float*) alloc((size_t)ROWS_BE * H_ * 4);
  __bf16* aggb  = (__bf16*)alloc((size_t)ROWS_BN * H_ * 2);
  __bf16* h1b   = (__bf16*)alloc((size_t)ROWS_BN * H_ * 2);
  float*  h2f   = (float*) alloc((size_t)ROWS_BN * H_ * 4);

  // Weight packing (once per launch; deterministic)
  k_pack<<<512, 256, 0, stream>>>(msg_w1 + 514 * H_, Wp1, 512);  // type-1, rows 0..511
  k_pack<<<256, 256, 0, stream>>>(msg_w2 + H_ * H_, Wp2, H_);    // type-1
  k_pack<<<256, 256, 0, stream>>>(w_hr, Wphr, H_);
  k_pack<<<256, 256, 0, stream>>>(w_hi, Wphi, H_);
  k_pack<<<256, 256, 0, stream>>>(w_hn, Wphn, H_);
  k_pack<<<256, 256, 0, stream>>>(w_o1, Wpo1, H_);
  k_pack<<<256, 256, 0, stream>>>(w_o2, Wpo2, H_);
  k_init<<<800, 256, 0, stream>>>(inputs, hid, hidb, prev);

  const int gBig = (E_ * 4) / 8;            // 1225 blocks, 8 waves/block
  const int gSm  = (ROWS_BN / 16) * 16 / 8; // 100 blocks
  for (int t = 0; t < T_ - 1; ++t) {
    k_msg1<<<gBig, 256, 0, stream>>>(hidb, Wp1, msg_w1, msg_b1, weather, t, msg1);
    k_msg2<<<gBig, 256, 0, stream>>>(msg1, Wp2, msg_b2, rel_types, am);
    k_agg <<<800, 256, 0, stream>>>(am, aggb);
    k_gru <<<gSm, 256, 0, stream>>>(aggb, Wphr, Wphi, Wphn, w_ir, b_ir, w_ii, b_ii,
                                    w_in, b_in, inputs, weather, prev, burnp, t,
                                    hid, hidb);
    k_mlp <<<gSm, 256, 0, stream>>>(hidb, Wpo1, b_o1, h1b, (float*)nullptr);
    k_mlp <<<gSm, 256, 0, stream>>>(h1b, Wpo2, b_o2, (__bf16*)nullptr, h2f);
    k_pred<<<4, 256, 0, stream>>>(h2f, w_o3, b_o3, inputs, burnp, t, prev,
                                  (float*)d_out);
  }
}
